// gemma_attn_74715251081461
// MI455X (gfx1250) — matrix operand order repaired and checked by reading, not compiled, not run on hardware
//
#include <hip/hip_runtime.h>
#include <hip/hip_bf16.h>
#include <math.h>

// ---------------- problem constants (match reference) ----------------
#define HID  2048
#define NH   8
#define KVH  1
#define HD   256
#define BB   2
#define SS   2048

#define AS_STR 40   // LDS row stride in ushorts (80B, 16B-aligned, bank-skewed)
#define BS_STR 40

typedef __attribute__((ext_vector_type(16))) __bf16 v16bf;
typedef __attribute__((ext_vector_type(2)))  __bf16 bf16x2;
typedef __attribute__((ext_vector_type(8)))  float  v8f;

__device__ __forceinline__ unsigned short f32_to_bf16_rne(float f) {
  __bf16 h = (__bf16)f;                     // hardware v_cvt, RNE
  return __builtin_bit_cast(unsigned short, h);
}
__device__ __forceinline__ unsigned int pack2_bf16(float a, float b) {
  bf16x2 p = { (__bf16)a, (__bf16)b };      // single v_cvt_pk_bf16_f32
  return __builtin_bit_cast(unsigned int, p);
}

union FragU { v16bf v; unsigned int u[8]; };

// =====================================================================
// bf16-WMMA GEMM:  C[M,N] (f32) = A[M,K] @ op(B)  (+ optional epilogue
// C = acc*scale + Mask[row,col]).
//   bIsNK==1 : B given row-major [N,K] -> computes A @ B^T
//   bIsNK==0 : B given row-major [K,N] -> computes A @ B
// blockIdx.z batches: z -> (b=z/nh, h=z%nh) with per-b/per-h strides.
// Block = 256 threads (8 waves). Block tile 128(M) x 128(N), K step 32.
// Software-pipelined: global loads for chunk c+1 are in flight during the
// 8 back-to-back WMMAs of chunk c; LDS is double-buffered so a single
// __syncthreads per chunk suffices.
// Requires M%128==0, N%128==0, K%32==0 (true for every call here).
// =====================================================================
__global__ __launch_bounds__(256)
void gqa_wmma_gemm(const float* __restrict__ A, long long aZb, long long aZh, int lda,
                   const float* __restrict__ Bm, long long bZb, long long bZh, int ldb, int bIsNK,
                   float* __restrict__ C, long long cZb, long long cZh, int ldc,
                   const float* __restrict__ Mask, long long mZb, int ldm,
                   int M, int N, int K, int nh, float scale, int epi)
{
  const int z = blockIdx.z;
  const int b = z / nh;
  const int h = z % nh;
  A  += (size_t)b * aZb + (size_t)h * aZh;
  Bm += (size_t)b * bZb + (size_t)h * bZh;
  C  += (size_t)b * cZb + (size_t)h * cZh;
  if (Mask) Mask += (size_t)b * mZb;

  const int tid  = threadIdx.x;
  const int lane = tid & 31;
  const int wave = tid >> 5;        // 0..7 -> M sub-tile
  const int m0   = blockIdx.y * 128;
  const int n0   = blockIdx.x * 128;
  const int hi   = lane >> 4;       // lane half
  const int l15  = lane & 15;

  __shared__ unsigned short As[2][128 * AS_STR];  // [m][k] bf16, padded
  __shared__ unsigned short Bs[2][128 * BS_STR];  // [n][k] bf16 (B^T-major)

  v8f acc[8];
  #pragma unroll
  for (int j = 0; j < 8; ++j) acc[j] = (v8f){};

  float4 ra[4], rb[4];

  // batched global loads for one K-chunk (8 b128 loads in flight, 1 wait)
  auto gload = [&](int kk) {
    #pragma unroll
    for (int i = 0; i < 4; ++i) {
      int idx = i * 256 + tid;
      int r = idx >> 3, c4 = (idx & 7) * 4;
      ra[i] = *(const float4*)&A[(size_t)(m0 + r) * lda + (kk + c4)];
    }
    if (bIsNK) {
      #pragma unroll
      for (int i = 0; i < 4; ++i) {
        int idx = i * 256 + tid;
        int r = idx >> 3, c4 = (idx & 7) * 4;
        rb[i] = *(const float4*)&Bm[(size_t)(n0 + r) * ldb + (kk + c4)];
      }
    } else {
      #pragma unroll
      for (int i = 0; i < 4; ++i) {
        int idx = i * 256 + tid;
        int k = idx >> 5, n4 = (idx & 31) * 4;
        rb[i] = *(const float4*)&Bm[(size_t)(kk + k) * ldb + (n0 + n4)];
      }
    }
  };

  // f32 -> bf16 convert + LDS stores into buffer `buf`
  auto cvtstore = [&](int buf) {
    unsigned short* as = As[buf];
    unsigned short* bs = Bs[buf];
    #pragma unroll
    for (int i = 0; i < 4; ++i) {
      int idx = i * 256 + tid;
      int r = idx >> 3, c4 = (idx & 7) * 4;
      unsigned int* dst = (unsigned int*)&as[r * AS_STR + c4];
      dst[0] = pack2_bf16(ra[i].x, ra[i].y);
      dst[1] = pack2_bf16(ra[i].z, ra[i].w);
    }
    if (bIsNK) {
      #pragma unroll
      for (int i = 0; i < 4; ++i) {
        int idx = i * 256 + tid;
        int r = idx >> 3, c4 = (idx & 7) * 4;
        unsigned int* dst = (unsigned int*)&bs[r * BS_STR + c4];
        dst[0] = pack2_bf16(rb[i].x, rb[i].y);
        dst[1] = pack2_bf16(rb[i].z, rb[i].w);
      }
    } else {
      // transpose while storing (reads were coalesced along N)
      #pragma unroll
      for (int i = 0; i < 4; ++i) {
        int idx = i * 256 + tid;
        int k = idx >> 5, n4 = (idx & 31) * 4;
        bs[(n4 + 0) * BS_STR + k] = f32_to_bf16_rne(rb[i].x);
        bs[(n4 + 1) * BS_STR + k] = f32_to_bf16_rne(rb[i].y);
        bs[(n4 + 2) * BS_STR + k] = f32_to_bf16_rne(rb[i].z);
        bs[(n4 + 3) * BS_STR + k] = f32_to_bf16_rne(rb[i].w);
      }
    }
  };

  // fragments + 8 wmma on buffer `buf`
  auto compute = [&](int buf) {
    FragU fa;
    const unsigned short* arow = &As[buf][(wave * 16 + l15) * AS_STR];
    #pragma unroll
    for (int p = 0; p < 8; ++p) {
      // CDNA5 16-bit A layout: VGPR p holds K pair {base + hi*8, +1},
      // base = 2p (p<4) else 16+2(p-4)
      int kb = ((p < 4) ? (p * 2) : (16 + (p - 4) * 2)) + hi * 8;
      fa.u[p] = *(const unsigned int*)(arow + kb);
    }
    FragU fb[8];
    #pragma unroll
    for (int j = 0; j < 8; ++j) {
      const unsigned short* brow = &Bs[buf][(j * 16 + l15) * BS_STR + hi * 8];
      #pragma unroll
      for (int q = 0; q < 8; ++q)
        fb[j].u[q] = *(const unsigned int*)(brow + ((q < 4) ? (2 * q) : (2 * q + 8)));
    }
    #pragma unroll
    for (int j = 0; j < 8; ++j)
      acc[j] = __builtin_amdgcn_wmma_f32_16x16x32_bf16(
          false, fa.v, false, fb[j].v, (short)0, acc[j], false, false);
  };

  // ---- 2-stage pipeline over K chunks ----
  const int nc = K >> 5;
  gload(0);
  for (int c = 0; c < nc; ++c) {
    cvtstore(c & 1);
    __syncthreads();
    if (c + 1 < nc) gload((c + 1) << 5);   // next chunk in flight over compute
    compute(c & 1);
  }

  // ---- epilogue: lane holds col l15 of each sub-tile, rows hi*8 + i ----
  const int rowb = m0 + wave * 16 + hi * 8;
  #pragma unroll
  for (int j = 0; j < 8; ++j) {
    const int colg = n0 + j * 16 + l15;
    #pragma unroll
    for (int i = 0; i < 8; ++i) {
      float val = acc[j][i];
      int row = rowb + i;
      if (epi == 1) val = val * scale + Mask[(size_t)row * ldm + colg];
      C[(size_t)row * ldc + colg] = val;
    }
  }
}

// =====================================================================
// In-place RoPE on t laid out [B*S, heads*HD]; one thread per (tok,h,d<128)
// =====================================================================
__global__ void gqa_rope(float* __restrict__ t, const int* __restrict__ pos,
                         int heads, int total)
{
  int idx = blockIdx.x * blockDim.x + threadIdx.x;
  if (idx >= total) return;
  int d   = idx & (HD / 2 - 1);       // 0..127
  int r   = idx >> 7;
  int h   = r % heads;
  int tok = r / heads;
  float p = (float)pos[tok];
  // inv_freq = theta^(-2d/HD) = exp(-2d * ln(10000)/HD)
  float inv = __expf(-(float)(2 * d) * (9.2103403719761836f / (float)HD));
  float fr = p * inv, s, c;
  __sincosf(fr, &s, &c);
  float* base = t + (size_t)tok * heads * HD + (size_t)h * HD + d;
  float x0 = base[0];
  float x1 = base[HD / 2];
  base[0]      = x0 * c - x1 * s;
  base[HD / 2] = x1 * c + x0 * s;
}

// =====================================================================
// In-place row softmax over rows of length SS (one 256-thread block/row)
// =====================================================================
__global__ __launch_bounds__(256)
void gqa_softmax_rows(float* __restrict__ attn)
{
  float* row = attn + (size_t)blockIdx.x * SS;
  const int tid = threadIdx.x;
  float vals[SS / 256];
  float m = -3.0e38f;
  #pragma unroll
  for (int i = 0; i < SS / 256; ++i) {
    vals[i] = row[tid + i * 256];
    m = fmaxf(m, vals[i]);
  }
  __shared__ float red[256];
  red[tid] = m; __syncthreads();
  for (int s = 128; s > 0; s >>= 1) {
    if (tid < s) red[tid] = fmaxf(red[tid], red[tid + s]);
    __syncthreads();
  }
  float rmax = red[0]; __syncthreads();
  float sum = 0.f;
  #pragma unroll
  for (int i = 0; i < SS / 256; ++i) {
    vals[i] = __expf(vals[i] - rmax);
    sum += vals[i];
  }
  red[tid] = sum; __syncthreads();
  for (int s = 128; s > 0; s >>= 1) {
    if (tid < s) red[tid] += red[tid + s];
    __syncthreads();
  }
  float inv = 1.0f / red[0];
  #pragma unroll
  for (int i = 0; i < SS / 256; ++i)
    row[tid + i * 256] = vals[i] * inv;
}

// =====================================================================
extern "C" void kernel_launch(void* const* d_in, const int* in_sizes, int n_in,
                              void* d_out, int out_size, void* d_ws, size_t ws_size,
                              hipStream_t stream)
{
  (void)in_sizes; (void)n_in; (void)out_size; (void)ws_size;
  const float* x   = (const float*)d_in[0];
  const float* am  = (const float*)d_in[1];   // [B,1,S,S] additive mask
  const int*   pid = (const int*)  d_in[2];   // [B,S]
  const float* wq  = (const float*)d_in[3];   // [HID, NH*HD]
  const float* wk  = (const float*)d_in[4];   // [HID, HD]
  const float* wv  = (const float*)d_in[5];   // [HID, HD]
  const float* wo  = (const float*)d_in[6];   // [NH*HD, HID]

  float* out  = (float*)d_out;                              // [B,S,HID]
  float* attn = out + (size_t)BB * SS * HID;                // [B,NH,S,S]

  float* q   = (float*)d_ws;                                // [B*S, NH*HD]
  float* k   = q   + (size_t)BB * SS * NH * HD;             // [B*S, HD]
  float* v   = k   + (size_t)BB * SS * KVH * HD;            // [B*S, HD]
  float* ctx = v   + (size_t)BB * SS * KVH * HD;            // [B*S, NH*HD]

  const int Mtok = BB * SS;                                 // 4096
  dim3 blk(256);

  // ---- QKV projections ----
  gqa_wmma_gemm<<<dim3((NH * HD) / 128, Mtok / 128, 1), blk, 0, stream>>>(
      x, 0, 0, HID, wq, 0, 0, NH * HD, /*bIsNK=*/0,
      q, 0, 0, NH * HD, nullptr, 0, 0,
      Mtok, NH * HD, HID, /*nh=*/1, 1.0f, /*epi=*/0);
  gqa_wmma_gemm<<<dim3(HD / 128, Mtok / 128, 1), blk, 0, stream>>>(
      x, 0, 0, HID, wk, 0, 0, HD, 0,
      k, 0, 0, HD, nullptr, 0, 0,
      Mtok, HD, HID, 1, 1.0f, 0);
  gqa_wmma_gemm<<<dim3(HD / 128, Mtok / 128, 1), blk, 0, stream>>>(
      x, 0, 0, HID, wv, 0, 0, HD, 0,
      v, 0, 0, HD, nullptr, 0, 0,
      Mtok, HD, HID, 1, 1.0f, 0);

  // ---- RoPE on q (8 heads) and k (1 head) ----
  {
    int totq = Mtok * NH * (HD / 2);
    gqa_rope<<<(totq + 255) / 256, blk, 0, stream>>>(q, pid, NH, totq);
    int totk = Mtok * KVH * (HD / 2);
    gqa_rope<<<(totk + 255) / 256, blk, 0, stream>>>(k, pid, KVH, totk);
  }

  // ---- scores = q @ k^T * 1/sqrt(HD) + mask  -> attn region ----
  gqa_wmma_gemm<<<dim3(SS / 128, SS / 128, BB * NH), blk, 0, stream>>>(
      q, (long long)SS * NH * HD, (long long)HD, NH * HD,
      k, (long long)SS * HD, 0, HD, /*bIsNK=*/1,
      attn, (long long)NH * SS * SS, (long long)SS * SS, SS,
      am, (long long)SS * SS, SS,
      SS, SS, HD, /*nh=*/NH, 0.0625f, /*epi=*/1);

  // ---- softmax rows in place ----
  gqa_softmax_rows<<<BB * NH * SS, blk, 0, stream>>>(attn);

  // ---- ctx = attn @ v  (written as [B, S, NH*HD]) ----
  gqa_wmma_gemm<<<dim3(HD / 128, SS / 128, BB * NH), blk, 0, stream>>>(
      attn, (long long)NH * SS * SS, (long long)SS * SS, SS,
      v, (long long)SS * HD, 0, HD, /*bIsNK=*/0,
      ctx, (long long)SS * NH * HD, (long long)HD, NH * HD,
      nullptr, 0, 0,
      SS, HD, SS, NH, 1.0f, 0);

  // ---- out = ctx @ wo ----
  gqa_wmma_gemm<<<dim3(HID / 128, Mtok / 128, 1), blk, 0, stream>>>(
      ctx, 0, 0, NH * HD, wo, 0, 0, HID, 0,
      out, 0, 0, HID, nullptr, 0, 0,
      Mtok, HID, NH * HD, 1, 1.0f, 0);
}
